// SANet_4861902979710
// MI455X (gfx1250) — matrix operand order repaired and checked by reading, not compiled, not run on hardware
//
#include <hip/hip_runtime.h>

#define B_  4
#define C_  512
#define DH  128
#define N_  4096

typedef __attribute__((ext_vector_type(16))) __bf16 v16bf;
typedef __attribute__((ext_vector_type(8)))  __bf16 v8bf;
typedef __attribute__((ext_vector_type(8)))  float  v8f;

// ---------------------------------------------------------------------------
// Kernel 1: 1x1-conv projections.
//   qt[i*128 + d] = sum_c wq[d][c]*content[b][c][i] + bq[d]   (bf16, [B][N][D])
//   kt[i*128 + d] = same with wk/style                        (bf16, [B][N][D])
//   vv[c*N + i]   = sum_c' wv[c][c']*style[b][c'][i] + bv[c]  (bf16, [B][C][N])
// grid: (N/256, 768, B), block 256.  y<128 -> q, y<256 -> k, else v.
// ---------------------------------------------------------------------------
__global__ void proj_kernel(const float* __restrict__ content,
                            const float* __restrict__ style,
                            const float* __restrict__ wq, const float* __restrict__ bq,
                            const float* __restrict__ wk, const float* __restrict__ bk,
                            const float* __restrict__ wv, const float* __restrict__ bv,
                            __bf16* __restrict__ qt, __bf16* __restrict__ kt,
                            __bf16* __restrict__ vv) {
    const int i  = blockIdx.x * 256 + threadIdx.x;   // spatial index, lanes coalesce
    const int ch = blockIdx.y;                       // 0..767
    const int b  = blockIdx.z;

    const float* src;
    const float* w;
    float acc;
    int row;
    if (ch < 128)      { row = ch;       w = wq + row * C_; acc = bq[row]; src = content; }
    else if (ch < 256) { row = ch - 128; w = wk + row * C_; acc = bk[row]; src = style;   }
    else               { row = ch - 256; w = wv + row * C_; acc = bv[row]; src = style;   }
    src += (size_t)b * C_ * N_ + i;

#pragma unroll 8
    for (int c = 0; c < C_; ++c)
        acc += w[c] * src[(size_t)c * N_];   // w uniform -> scalar loads; src coalesced

    const __bf16 o = (__bf16)acc;
    if (ch < 128)      qt[((size_t)b * N_ + i) * DH + row] = o;
    else if (ch < 256) kt[((size_t)b * N_ + i) * DH + row] = o;
    else               vv[((size_t)b * C_ + row) * N_ + i] = o;   // coalesced
}

// A-fragment (16x32 bf16, lane m holds row m, K = hi*16 + {0..15}) from a
// K-contiguous row: two 16-byte loads + shuffle.
__device__ __forceinline__ v16bf load_a_frag(const __bf16* rowp) {
    v8bf lo = *(const v8bf*)(rowp);        // K offsets +0..+7
    v8bf hi = *(const v8bf*)(rowp + 8);    // K offsets +8..+15
    return __builtin_shufflevector(lo, hi, 0,1,2,3,4,5,6,7,8,9,10,11,12,13,14,15);
}

// ---------------------------------------------------------------------------
// Kernel 2: S[i][j] = sum_d qt[i][d] * kt[j][d]   (f32 scores, [N][N])
// grid: (N/16, 16), block 256 = 8 waves; wave w owns j-tiles {base, base+16},
// sharing one A fragment set (4 K-steps over D=128, 8 WMMAs/wave).
// ---------------------------------------------------------------------------
__global__ void energy_kernel(const __bf16* __restrict__ qt,   // [N][128]
                              const __bf16* __restrict__ kt,   // [N][128]
                              float* __restrict__ S) {         // [N][N]
    const int lane = threadIdx.x & 31;
    const int wave = threadIdx.x >> 5;
    const int i0   = blockIdx.x * 16;
    const int j0   = (blockIdx.y * 8 + wave) * 32;
    const int m    = lane & 15;
    const int hi   = lane >> 4;

    const __bf16* arow  = qt + (size_t)(i0 + m) * DH + hi * 16;
    const __bf16* brow0 = kt + (size_t)(j0 + m) * DH + hi * 16;
    const __bf16* brow1 = brow0 + (size_t)16 * DH;

    v8f c0 = {};
    v8f c1 = {};
#pragma unroll
    for (int kk = 0; kk < 4; ++kk) {
        v16bf a  = load_a_frag(arow + kk * 32);
        v16bf b0 = *(const v16bf*)(brow0 + kk * 32);   // B: K = kk*32 + hi*16 + e
        v16bf b1 = *(const v16bf*)(brow1 + kk * 32);
        c0 = __builtin_amdgcn_wmma_f32_16x16x32_bf16(false, a, false, b0,
                                                     (short)0, c0, false, false);
        c1 = __builtin_amdgcn_wmma_f32_16x16x32_bf16(false, a, false, b1,
                                                     (short)0, c1, false, false);
    }

    // C/D layout: VGPR r -> M = r + hi*8, N = lane&15
    float* srow = S + (size_t)(i0 + hi * 8) * N_ + j0 + m;
#pragma unroll
    for (int r = 0; r < 8; ++r) {
        srow[(size_t)r * N_]      = c0[r];
        srow[(size_t)r * N_ + 16] = c1[r];
    }
}

// ---------------------------------------------------------------------------
// Kernel 3: row softmax with post-softmax 1/sqrt(C) folded in.
//   P[i][j] = exp(S[i][j]-max)/sum * (1/sqrt(512))   (bf16, [N][N])
// grid: N blocks of 256.  Row cached in LDS (16KB): S read once from global.
// ---------------------------------------------------------------------------
__global__ void softmax_kernel(const float* __restrict__ S, __bf16* __restrict__ P) {
    const int i = blockIdx.x;
    const int t = threadIdx.x;
    const float* row = S + (size_t)i * N_;
    __shared__ float ex[N_];      // 16 KB row cache
    __shared__ float red[256];

    float m = -3.402823466e38f;
    for (int j = t; j < N_; j += 256) {
        float v = row[j];
        ex[j] = v;
        m = fmaxf(m, v);
    }
    red[t] = m; __syncthreads();
    for (int s = 128; s > 0; s >>= 1) {
        if (t < s) red[t] = fmaxf(red[t], red[t + s]);
        __syncthreads();
    }
    m = red[0]; __syncthreads();

    float sum = 0.f;
    for (int j = t; j < N_; j += 256) {
        float e = __expf(ex[j] - m);
        ex[j] = e;
        sum += e;
    }
    red[t] = sum; __syncthreads();
    for (int s = 128; s > 0; s >>= 1) {
        if (t < s) red[t] += red[t + s];
        __syncthreads();
    }
    const float scale = 0.0441941738241592f / red[0];   // (1/sqrt(512)) / sum

    __bf16* prow = P + (size_t)i * N_;
    for (int j = t; j < N_; j += 256) prow[j] = (__bf16)(ex[j] * scale);
}

// ---------------------------------------------------------------------------
// Kernel 4: out[c][i] = gamma * (sum_j P[i][j] * V[c][j]) + content[c][i]
// grid: N/64 blocks of 256.  Wave w owns 4 i-tiles {base+16t} x 4 c-tiles
// {w*64+16t}: each B fragment feeds 4 WMMAs (16 acc = 128 VGPRs,
// ~32 flops/byte), so only 64 blocks/batch stream V -> ~256MB/batch L2
// traffic.  128 WMMA K-steps over j; prefetch pulls the V stream ahead.
// ---------------------------------------------------------------------------
__global__ void __launch_bounds__(256, 1)
av_kernel(const __bf16* __restrict__ P,        // [N][N]
          const __bf16* __restrict__ V,        // [C][N]
          const float* __restrict__ content,   // [C][N] (batch base)
          const float* __restrict__ gamma,
          float* __restrict__ out) {           // [C][N] (batch base)
    const int lane  = threadIdx.x & 31;
    const int wave  = threadIdx.x >> 5;
    const int ibase = blockIdx.x * 64;
    const int m     = lane & 15;
    const int hi    = lane >> 4;
    const int c0    = wave * 64;

    const __bf16* prow  = P + (size_t)(ibase + m) * N_ + hi * 16;  // i-tile stride 16*N
    const __bf16* vbase = V + (size_t)(c0 + m) * N_ + hi * 16;     // c-tile stride 16*N

    v8f acc[4][4];
#pragma unroll
    for (int it = 0; it < 4; ++it)
#pragma unroll
        for (int ct = 0; ct < 4; ++ct)
            acc[it][ct] = (v8f){0.f,0.f,0.f,0.f,0.f,0.f,0.f,0.f};

#pragma unroll 1
    for (int kk = 0; kk < N_ / 32; ++kk) {
        // prefetch next V K-window (lowers to global_prefetch_b8)
        __builtin_prefetch((const void*)(vbase + (size_t)(kk + 4) * 32), 0, 0);

        v16bf a[4];
#pragma unroll
        for (int it = 0; it < 4; ++it)
            a[it] = load_a_frag(prow + (size_t)it * 16 * N_ + kk * 32);

#pragma unroll
        for (int ct = 0; ct < 4; ++ct) {
            v16bf bf = *(const v16bf*)(vbase + (size_t)ct * 16 * N_ + kk * 32);
#pragma unroll
            for (int it = 0; it < 4; ++it)
                acc[it][ct] = __builtin_amdgcn_wmma_f32_16x16x32_bf16(
                    false, a[it], false, bf, (short)0, acc[it][ct], false, false);
        }
    }

    const float g = gamma[0];
#pragma unroll
    for (int it = 0; it < 4; ++it) {
#pragma unroll
        for (int ct = 0; ct < 4; ++ct) {
            const int cc = c0 + ct * 16 + m;                // output channel
            const int ii = ibase + it * 16 + hi * 8;        // spatial base
            float* optr       = out     + (size_t)cc * N_ + ii;
            const float* cptr = content + (size_t)cc * N_ + ii;
#pragma unroll
            for (int r = 0; r < 8; ++r)                     // 32B contiguous per lane
                optr[r] = g * acc[it][ct][r] + cptr[r];
        }
    }
}

// ---------------------------------------------------------------------------
extern "C" void kernel_launch(void* const* d_in, const int* in_sizes, int n_in,
                              void* d_out, int out_size, void* d_ws, size_t ws_size,
                              hipStream_t stream) {
    const float* content = (const float*)d_in[0];
    const float* style   = (const float*)d_in[1];
    const float* wq      = (const float*)d_in[2];
    const float* bq      = (const float*)d_in[3];
    const float* wk      = (const float*)d_in[4];
    const float* bk      = (const float*)d_in[5];
    const float* wv      = (const float*)d_in[6];
    const float* bv      = (const float*)d_in[7];
    const float* gamma   = (const float*)d_in[8];
    float* out           = (float*)d_out;

    // Workspace layout (120 MB total):
    //   [0,4MB)    QT  bf16 [B][N][128]
    //   [4,8MB)    KT  bf16 [B][N][128]
    //   [8,24MB)   V   bf16 [B][C][N]
    //   [24,88MB)  S   f32  [N][N]      (reused per batch)
    //   [88,120MB) P   bf16 [N][N]      (reused per batch)
    char* ws = (char*)d_ws;
    __bf16* qt = (__bf16*)(ws);
    __bf16* kt = (__bf16*)(ws + ((size_t)4  << 20));
    __bf16* vv = (__bf16*)(ws + ((size_t)8  << 20));
    float*  S  = (float*) (ws + ((size_t)24 << 20));
    __bf16* P  = (__bf16*)(ws + ((size_t)88 << 20));

    proj_kernel<<<dim3(N_ / 256, 768, B_), 256, 0, stream>>>(
        content, style, wq, bq, wk, bk, wv, bv, qt, kt, vv);

    for (int b = 0; b < B_; ++b) {
        energy_kernel<<<dim3(N_ / 16, 16), 256, 0, stream>>>(
            qt + (size_t)b * N_ * DH, kt + (size_t)b * N_ * DH, S);
        softmax_kernel<<<N_, 256, 0, stream>>>(S, P);
        av_kernel<<<N_ / 64, 256, 0, stream>>>(
            P, vv + (size_t)b * C_ * N_, content + (size_t)b * C_ * N_,
            gamma, out + (size_t)b * C_ * N_);
    }
}
